// A3TGCN2_network_90305982366080
// MI455X (gfx1250) — matrix operand order repaired and checked by reading, not compiled, not run on hardware
//
#include <hip/hip_runtime.h>

// ---------------------------------------------------------------------------
// A3TGCN2: B=32, N=51, F=4, P=137, C=256, E=600. M = B*N = 1632 rows.
// Hot loop: per timestep, three [1632,512]@[512,256] GEMMs on WMMA f16->f32.
// Each wave computes TWO 16x16 N-tiles: one A-fragment load feeds two
// independent v_wmma chains (better XDL occupancy, half the A traffic).
// ---------------------------------------------------------------------------

#define Bc 32
#define Nc 51
#define Fc 4
#define Pc 137
#define Cc 256
#define Ec 600
#define Mr 1632          // B*N
#define K2 512           // 2*C
#define NE 651           // E + N self loops

typedef _Float16 v16h __attribute__((ext_vector_type(16)));
typedef _Float16 v8h  __attribute__((ext_vector_type(8)));
typedef float    v8f  __attribute__((ext_vector_type(8)));

// ---------------- graph prep: dense normalized adjacency --------------------

__global__ void k_zero_small(float* Amat, float* deg) {
    int i = blockIdx.x * 256 + threadIdx.x;
    if (i < Nc * Nc) Amat[i] = 0.f;
    if (i < Nc) deg[i] = 0.f;
}

__global__ void k_add_deg(const int* __restrict__ ei, float* deg) {
    int e = blockIdx.x * 256 + threadIdx.x;
    if (e >= NE) return;
    int d = (e < Ec) ? ei[Ec + e] : (e - Ec);
    atomicAdd(&deg[d], 1.0f);
}

__global__ void k_build_A(const int* __restrict__ ei, const float* __restrict__ deg,
                          float* Amat) {
    int e = blockIdx.x * 256 + threadIdx.x;
    if (e >= NE) return;
    int s = (e < Ec) ? ei[e] : (e - Ec);
    int d = (e < Ec) ? ei[Ec + e] : (e - Ec);
    float ds = deg[s] > 0.f ? rsqrtf(deg[s]) : 0.f;
    float dd = deg[d] > 0.f ? rsqrtf(deg[d]) : 0.f;
    atomicAdd(&Amat[d * Nc + s], ds * dd);
}

// ---------------- softmax over attention (P=137) ----------------------------

__global__ void k_softmax(const float* __restrict__ att, float* probs) {
    __shared__ float sm[256];
    int i = threadIdx.x;
    float v = (i < Pc) ? att[i] : -1e30f;
    sm[i] = v; __syncthreads();
    for (int s = 128; s > 0; s >>= 1) { if (i < s) sm[i] = fmaxf(sm[i], sm[i + s]); __syncthreads(); }
    float mx = sm[0]; __syncthreads();
    float e = (i < Pc) ? __expf(v - mx) : 0.f;
    sm[i] = e; __syncthreads();
    for (int s = 128; s > 0; s >>= 1) { if (i < s) sm[i] += sm[i + s]; __syncthreads(); }
    if (i < Pc) probs[i] = e / sm[0];
}

// ---------------- Y[t,row,f] = sum_m Ahat[n,m] * x[b,m,f,t] -----------------

__global__ void k_build_Y(const float* __restrict__ x, const float* __restrict__ Amat,
                          float* __restrict__ Y) {
    int idx = blockIdx.x * 256 + threadIdx.x;
    if (idx >= Pc * Mr * Fc) return;
    int f   = idx & 3;
    int rem = idx >> 2;          // t*Mr + row
    int row = rem % Mr;
    int t   = rem / Mr;
    int n = row % Nc, b = row / Nc;
    float s = 0.f;
    const float* arow = Amat + n * Nc;
    for (int m = 0; m < Nc; ++m)
        s += arow[m] * x[((b * Nc + m) * Fc + f) * Pc + t];
    Y[idx] = s;
}

// ---------------- w_lin transpose to f16, [n][k] layout ---------------------

__global__ void k_wt_lin(const float* __restrict__ wz, const float* __restrict__ wr,
                         const float* __restrict__ wh,
                         _Float16* WTz, _Float16* WTr, _Float16* WTh) {
    int idx = blockIdx.x * 256 + threadIdx.x;   // over Cc*K2
    if (idx >= Cc * K2) return;
    int n = idx / K2, k = idx % K2;
    WTz[idx] = (_Float16)wz[k * Cc + n];
    WTr[idx] = (_Float16)wr[k * Cc + n];
    WTh[idx] = (_Float16)wh[k * Cc + n];
}

// ---------------- init: H=0, Hacc=0, f16 H-halves of A_z/A_r = 0 ------------

__global__ void k_init_state(float* H, float* Hacc, _Float16* AzH, _Float16* ArH) {
    int idx = blockIdx.x * 256 + threadIdx.x;   // Mr*Cc
    int row = idx >> 8, c = idx & 255;
    H[idx] = 0.f; Hacc[idx] = 0.f;
    AzH[row * K2 + c] = (_Float16)0.f;
    ArH[row * K2 + c] = (_Float16)0.f;
}

// ---------------- per-step GCN: conv_g = Y[t] @ w_conv_g + b_conv_g ---------

__global__ __launch_bounds__(256)
void k_conv(const float* __restrict__ Y, int t,
            const float* __restrict__ wz, const float* __restrict__ bz,
            const float* __restrict__ wr, const float* __restrict__ br,
            const float* __restrict__ wh, const float* __restrict__ bh,
            _Float16* __restrict__ Az, _Float16* __restrict__ Ar,
            _Float16* __restrict__ Ah) {
    int row = blockIdx.x;
    int c   = threadIdx.x;
    const float* y = Y + (t * Mr + row) * Fc;
    float y0 = y[0], y1 = y[1], y2 = y[2], y3 = y[3];
    float cz = bz[c] + y0 * wz[c] + y1 * wz[Cc + c] + y2 * wz[2 * Cc + c] + y3 * wz[3 * Cc + c];
    float cr = br[c] + y0 * wr[c] + y1 * wr[Cc + c] + y2 * wr[2 * Cc + c] + y3 * wr[3 * Cc + c];
    float ch = bh[c] + y0 * wh[c] + y1 * wh[Cc + c] + y2 * wh[2 * Cc + c] + y3 * wh[3 * Cc + c];
    Az[row * K2 + c] = (_Float16)cz;
    Ar[row * K2 + c] = (_Float16)cr;
    Ah[row * K2 + c] = (_Float16)ch;
}

// ---------------- WMMA GEMM [1632,512] @ [512,256] with fused gate epilogue -
// Each wave: one 16-row strip x two adjacent 16-col tiles, K=512 in 16 steps.
// mode 0: Z = sigmoid(.)            -> Zbuf
// mode 1: R = sigmoid(.)            -> A_h HR-half = f16(H * R)
// mode 2: Ht = tanh(.); Hnew = Z*H + (1-Z)*Ht; Hacc += p[t]*Hnew;
//         H = Hnew; A_z/A_r H-halves = f16(Hnew)

__global__ __launch_bounds__(256)
void k_gemm_gate(const _Float16* __restrict__ Abuf,
                 const _Float16* __restrict__ WT,
                 const float* __restrict__ bias,
                 int mode, int t,
                 const float* __restrict__ probs,
                 float* __restrict__ Zbuf,
                 float* __restrict__ Hf32,
                 float* __restrict__ Hacc,
                 _Float16* __restrict__ AhHR,
                 _Float16* __restrict__ AzH,
                 _Float16* __restrict__ ArH) {
    const int lane = threadIdx.x & 31;
    const int wave = threadIdx.x >> 5;
    const int mt   = blockIdx.x;                 // 0..101 (M tiles)
    const int nt0  = wave * 2;                   // 0,2,..,14 (pair of N tiles)
    const int half = lane >> 4;                  // 0/1
    const int l16  = lane & 15;

    // A fragment base: row = mt*16 + l16; 16 contiguous K per lane group.
    const _Float16* aP  = Abuf + (size_t)(mt * 16 + l16) * K2 + half * 16;
    // B fragment bases: col = nt*16 + l16; 16 contiguous K per lane group.
    const _Float16* bP0 = WT + (size_t)(nt0 * 16 + l16) * K2 + half * 16;
    const _Float16* bP1 = bP0 + (size_t)16 * K2;

    v8f acc0 = {0.f, 0.f, 0.f, 0.f, 0.f, 0.f, 0.f, 0.f};
    v8f acc1 = {0.f, 0.f, 0.f, 0.f, 0.f, 0.f, 0.f, 0.f};

    #pragma unroll 4
    for (int kk = 0; kk < 16; ++kk) {
        v8h alo = *(const v8h*)(aP + kk * 32);
        v8h ahi = *(const v8h*)(aP + kk * 32 + 8);
        v16h b0 = *(const v16h*)(bP0 + kk * 32);
        v16h b1 = *(const v16h*)(bP1 + kk * 32);
        v16h a;
        #pragma unroll
        for (int i = 0; i < 8; ++i) { a[i] = alo[i]; a[i + 8] = ahi[i]; }
        if (kk + 1 < 16)
            __builtin_prefetch((const void*)(aP + (kk + 1) * 32), 0, 1);
        acc0 = __builtin_amdgcn_wmma_f32_16x16x32_f16(
            false, a, false, b0, (short)0, acc0, false, false);
        acc1 = __builtin_amdgcn_wmma_f32_16x16x32_f16(
            false, a, false, b1, (short)0, acc1, false, false);
    }

    const float pt = (mode == 2) ? probs[t] : 0.f;

    #pragma unroll
    for (int tile = 0; tile < 2; ++tile) {
        v8f acc = tile ? acc1 : acc0;
        const int col = (nt0 + tile) * 16 + l16;
        const float bv = bias[col];
        #pragma unroll
        for (int j = 0; j < 8; ++j) {
            int row = mt * 16 + half * 8 + j;
            int idx = row * Cc + col;
            float v = acc[j] + bv;
            if (mode == 0) {
                Zbuf[idx] = 1.f / (1.f + __expf(-v));
            } else if (mode == 1) {
                float r = 1.f / (1.f + __expf(-v));
                AhHR[(size_t)row * K2 + col] = (_Float16)(Hf32[idx] * r);
            } else {
                float ht = tanhf(v);
                float z  = Zbuf[idx];
                float h  = Hf32[idx];
                float hn = z * h + (1.f - z) * ht;
                Hf32[idx] = hn;
                Hacc[idx] += pt * hn;
                _Float16 hh = (_Float16)hn;
                AzH[(size_t)row * K2 + col] = hh;
                ArH[(size_t)row * K2 + col] = hh;
            }
        }
    }
}

// ---------------- head ------------------------------------------------------

__global__ __launch_bounds__(128)
void k_head1(const float* __restrict__ Hacc, const float* __restrict__ w1,
             const float* __restrict__ b1, const float* __restrict__ w3,
             const float* __restrict__ b3, float* __restrict__ t2) {
    __shared__ float sm[128];
    int row = blockIdx.x, j = threadIdx.x;
    float a = b1[j];
    const float* h = Hacc + row * Cc;
    for (int c = 0; c < Cc; ++c) a += h[c] * w1[c * 128 + j];
    sm[j] = w3[j] * a; __syncthreads();
    for (int s = 64; s > 0; s >>= 1) { if (j < s) sm[j] += sm[j + s]; __syncthreads(); }
    if (j == 0) t2[row] = sm[0] + b3[0];
}

__global__ __launch_bounds__(64)
void k_head2(const float* __restrict__ t2, const float* __restrict__ w4,
             const float* __restrict__ b4, float* __restrict__ out) {
    __shared__ float sm[64];
    int b = blockIdx.x, i = threadIdx.x;
    float v = (i < Nc) ? t2[b * Nc + i] * w4[i] : 0.f;
    sm[i] = v; __syncthreads();
    for (int s = 32; s > 0; s >>= 1) { if (i < s) sm[i] += sm[i + s]; __syncthreads(); }
    if (i == 0) out[b] = fmaxf(sm[0] + b4[0], 0.f);
}

// ---------------------------------------------------------------------------

extern "C" void kernel_launch(void* const* d_in, const int* in_sizes, int n_in,
                              void* d_out, int out_size, void* d_ws, size_t ws_size,
                              hipStream_t stream) {
    (void)in_sizes; (void)n_in; (void)out_size; (void)ws_size;

    const float* x        = (const float*)d_in[0];
    const int*   ei       = (const int*)  d_in[1];
    const float* w_conv_z = (const float*)d_in[2];
    const float* b_conv_z = (const float*)d_in[3];
    const float* w_conv_r = (const float*)d_in[4];
    const float* b_conv_r = (const float*)d_in[5];
    const float* w_conv_h = (const float*)d_in[6];
    const float* b_conv_h = (const float*)d_in[7];
    const float* w_lin_z  = (const float*)d_in[8];
    const float* b_lin_z  = (const float*)d_in[9];
    const float* w_lin_r  = (const float*)d_in[10];
    const float* b_lin_r  = (const float*)d_in[11];
    const float* w_lin_h  = (const float*)d_in[12];
    const float* b_lin_h  = (const float*)d_in[13];
    const float* att      = (const float*)d_in[14];
    const float* w1       = (const float*)d_in[15];
    const float* b1       = (const float*)d_in[16];
    const float* w3       = (const float*)d_in[17];
    const float* b3       = (const float*)d_in[18];
    const float* w4       = (const float*)d_in[19];
    const float* b4       = (const float*)d_in[20];
    float* out = (float*)d_out;

    // workspace carve (≈15 MB)
    char* p = (char*)d_ws;
    auto alloc = [&](size_t bytes) -> char* {
        char* r = p; p += (bytes + 255) & ~(size_t)255; return r;
    };
    float*    probs = (float*)   alloc(Pc * 4);
    float*    Amat  = (float*)   alloc(Nc * Nc * 4);
    float*    deg   = (float*)   alloc(Nc * 4);
    float*    Y     = (float*)   alloc((size_t)Pc * Mr * Fc * 4);
    _Float16* Az    = (_Float16*)alloc((size_t)Mr * K2 * 2);
    _Float16* Ar    = (_Float16*)alloc((size_t)Mr * K2 * 2);
    _Float16* Ah    = (_Float16*)alloc((size_t)Mr * K2 * 2);
    _Float16* WTz   = (_Float16*)alloc((size_t)Cc * K2 * 2);
    _Float16* WTr   = (_Float16*)alloc((size_t)Cc * K2 * 2);
    _Float16* WTh   = (_Float16*)alloc((size_t)Cc * K2 * 2);
    float*    Zbuf  = (float*)   alloc((size_t)Mr * Cc * 4);
    float*    Hf32  = (float*)   alloc((size_t)Mr * Cc * 4);
    float*    Hacc  = (float*)   alloc((size_t)Mr * Cc * 4);
    float*    t2    = (float*)   alloc(Mr * 4);

    // ---- prep (recomputed every call; deterministic) ----
    k_zero_small<<<(Nc * Nc + 255) / 256, 256, 0, stream>>>(Amat, deg);
    k_add_deg  <<<(NE + 255) / 256, 256, 0, stream>>>(ei, deg);
    k_build_A  <<<(NE + 255) / 256, 256, 0, stream>>>(ei, deg, Amat);
    k_softmax  <<<1, 256, 0, stream>>>(att, probs);
    k_build_Y  <<<(Pc * Mr * Fc + 255) / 256, 256, 0, stream>>>(x, Amat, Y);
    k_wt_lin   <<<(Cc * K2 + 255) / 256, 256, 0, stream>>>(w_lin_z, w_lin_r, w_lin_h,
                                                           WTz, WTr, WTh);
    k_init_state<<<Mr, 256, 0, stream>>>(Hf32, Hacc, Az + Cc, Ar + Cc);

    // ---- sequential GRU scan over P timesteps ----
    dim3 ggrid(Mr / 16);   // 102 M-tiles; 8 waves x 2 N-tiles cover N=256
    for (int t = 0; t < Pc; ++t) {
        k_conv<<<Mr, 256, 0, stream>>>(Y, t, w_conv_z, b_conv_z, w_conv_r, b_conv_r,
                                       w_conv_h, b_conv_h, Az, Ar, Ah);
        k_gemm_gate<<<ggrid, 256, 0, stream>>>(Az, WTz, b_lin_z, 0, t, probs,
                                               Zbuf, Hf32, Hacc, Ah + Cc, Az + Cc, Ar + Cc);
        k_gemm_gate<<<ggrid, 256, 0, stream>>>(Ar, WTr, b_lin_r, 1, t, probs,
                                               Zbuf, Hf32, Hacc, Ah + Cc, Az + Cc, Ar + Cc);
        k_gemm_gate<<<ggrid, 256, 0, stream>>>(Ah, WTh, b_lin_h, 2, t, probs,
                                               Zbuf, Hf32, Hacc, Ah + Cc, Az + Cc, Ar + Cc);
    }

    // ---- head ----
    k_head1<<<Mr, 128, 0, stream>>>(Hacc, w1, b1, w3, b3, t2);
    k_head2<<<Bc, 64, 0, stream>>>(t2, w4, b4, out);
}
